// MNEMatch_63660005261735
// MI455X (gfx1250) — matrix operand order repaired and checked by reading, not compiled, not run on hardware
//
#include <hip/hip_runtime.h>
#include <math.h>

// MNEMatch: B=128 pairs, N=256 nodes, D=384 dims.
// Phase 1: S = x1[b] @ x2[b]^T via v_wmma_f32_16x16x32_bf16, S resident in LDS.
// Phase 2: greedy max-weight matching entirely in LDS with incremental row-max cache.

typedef __attribute__((ext_vector_type(16))) __bf16 v16bf;
typedef __attribute__((ext_vector_type(8)))  float  v8f;

#define NN 256
#define DD 384
#define SSTRIDE 260   // padded LDS row stride (floats): 16B-aligned rows, bank-skewed

extern "C" __global__ __launch_bounds__(256, 1)
void mne_match_kernel(const float* __restrict__ x1, const float* __restrict__ x2,
                      float* __restrict__ out) {
  extern __shared__ float smem[];
  float* S      = smem;                         // NN * SSTRIDE floats (score matrix)
  float* rowmax = smem + NN * SSTRIDE;          // NN
  int*   rowarg = (int*)(rowmax + NN);          // NN
  int*   colA   = (int*)(rowarg + NN);          // NN (1 = column alive)
  float* redv   = (float*)(colA + NN);          // 8 per-wave partial max values
  int*   redr   = (int*)(redv + 8);             // 8 per-wave partial argmax rows
  int*   winC   = (int*)(redr + 8);             // 1 winning column broadcast

  const int b    = blockIdx.x;
  const int tid  = threadIdx.x;
  const int wave = tid >> 5;
  const int lane = tid & 31;
  const int half = lane >> 4;
  const int mn   = lane & 15;

  const float* X1 = x1 + (size_t)b * NN * DD;
  const float* X2 = x2 + (size_t)b * NN * DD;

  // ---------------- Phase 1: S = X1 * X2^T (bf16 WMMA, fp32 accumulate) -----
  // 256 tiles of 16x16; wave w owns tiles [w*32, w*32+32) => tm in {2w, 2w+1}.
  for (int t = 0; t < 32; ++t) {
    const int tile = wave * 32 + t;
    const int tm = tile >> 4;
    const int tn = tile & 15;
    const float* arow = X1 + (size_t)(tm * 16 + mn) * DD;  // lane's A row (M = mn)
    const float* brow = X2 + (size_t)(tn * 16 + mn) * DD;  // lane's B col  (N = mn)
    v8f acc = {};
    for (int ks = 0; ks < 12; ++ks) {
      const int kb = ks * 32;
      v16bf afrag, bfrag;
      #pragma unroll
      for (int v = 0; v < 8; ++v) {
        // A 16x32 bf16 layout: VGPR v holds K = (v<4 ? 2v : 16+2(v-4)) + 8*half, +1
        const int ka = kb + ((v < 4) ? (2 * v) : (16 + 2 * (v - 4))) + 8 * half;
        afrag[2 * v]     = (__bf16)arow[ka];
        afrag[2 * v + 1] = (__bf16)arow[ka + 1];
        // B 32x16 bf16 layout: VGPR v holds K = (v<4 ? 2v : 16+2(v-4)) + 8*half, +1 (N = lane%16)
        const int kc = kb + ((v < 4) ? (2 * v) : (16 + 2 * (v - 4))) + 8 * half;
        bfrag[2 * v]     = (__bf16)brow[kc];
        bfrag[2 * v + 1] = (__bf16)brow[kc + 1];
      }
      acc = __builtin_amdgcn_wmma_f32_16x16x32_bf16(
          /*neg_a=*/false, afrag, /*neg_b=*/false, bfrag,
          /*c_mod=*/(short)0, acc, /*reuse_a=*/false, /*reuse_b=*/false);
    }
    // C/D layout: VGPR j -> M = j + 8*half, N = lane%16
    #pragma unroll
    for (int j = 0; j < 8; ++j) {
      const int row = tm * 16 + j + 8 * half;
      const int col = tn * 16 + mn;
      S[row * SSTRIDE + col] = acc[j];
    }
  }
  colA[tid] = 1;
  __syncthreads();

  // ---------------- initial per-row maxima ----------------------------------
  {
    const float* Srow = S + tid * SSTRIDE;
    float best = -__builtin_inff();
    int   barg = 0;
    for (int c = 0; c < NN; ++c) {
      const float v = Srow[c];
      if (v > best) { best = v; barg = c; }   // first occurrence wins ties
    }
    rowmax[tid] = best;
    rowarg[tid] = barg;
  }
  __syncthreads();

  // ---------------- Phase 2: greedy matching, N picks -----------------------
  const float NEG_INF = -__builtin_inff();
  float total = 0.0f;  // meaningful on thread 0 only

  for (int it = 0; it < NN; ++it) {
    // argmax over rowmax[0..255]; tie-break = lowest row (flat-argmax semantics)
    float v = rowmax[tid];
    int   r = tid;
    #pragma unroll
    for (int off = 16; off > 0; off >>= 1) {
      const float ov = __shfl_xor(v, off, 32);
      const int   orr = __shfl_xor(r, off, 32);
      if (ov > v || (ov == v && orr < r)) { v = ov; r = orr; }
    }
    if (lane == 0) { redv[wave] = v; redr[wave] = r; }
    __syncthreads();
    if (tid == 0) {
      float bv = redv[0]; int br = redr[0];
      #pragma unroll
      for (int w = 1; w < 8; ++w) {
        if (redv[w] > bv || (redv[w] == bv && redr[w] < br)) { bv = redv[w]; br = redr[w]; }
      }
      const int bc = rowarg[br];
      total += bv;               // picked score S[br][bc]
      rowmax[br] = NEG_INF;      // kill row
      colA[bc]   = 0;            // kill column
      winC[0]    = bc;
    }
    __syncthreads();
    // rows whose cached argmax died must rescan (typically ~1 row)
    const int bc = winC[0];
    if (rowmax[tid] != NEG_INF && rowarg[tid] == bc) {
      const float* Srow = S + tid * SSTRIDE;
      float best = NEG_INF;
      int   barg = 0;
      for (int c = 0; c < NN; ++c) {
        if (colA[c]) {
          const float sv = Srow[c];
          if (sv > best) { best = sv; barg = c; }
        }
      }
      rowmax[tid] = best;
      rowarg[tid] = barg;
    }
    __syncthreads();
  }

  if (tid == 0) out[b] = tanhf(total / (float)NN);
}

extern "C" void kernel_launch(void* const* d_in, const int* in_sizes, int n_in,
                              void* d_out, int out_size, void* d_ws, size_t ws_size,
                              hipStream_t stream) {
  (void)in_sizes; (void)n_in; (void)d_ws; (void)ws_size; (void)out_size;
  const float* x1 = (const float*)d_in[0];
  const float* x2 = (const float*)d_in[1];
  float* out = (float*)d_out;

  // LDS: score matrix + matching state (~263 KB; CDNA5 WGP has 320 KB)
  const size_t shmem =
      (size_t)(NN * SSTRIDE + NN /*rowmax*/) * sizeof(float) +
      (size_t)(NN /*rowarg*/ + NN /*colA*/) * sizeof(int) +
      8 * sizeof(float) + 8 * sizeof(int) + sizeof(int);

  hipFuncSetAttribute((const void*)mne_match_kernel,
                      hipFuncAttributeMaxDynamicSharedMemorySize, (int)shmem);

  mne_match_kernel<<<dim3(128), dim3(256), shmem, stream>>>(x1, x2, out);
}
